// InferCellV2_2525440770352
// MI455X (gfx1250) — matrix operand order repaired and checked by reading, not compiled, not run on hardware
//
#include <hip/hip_runtime.h>

typedef __attribute__((ext_vector_type(16))) _Float16 v16h;
typedef __attribute__((ext_vector_type(8)))  _Float16 v8h;
typedef __attribute__((ext_vector_type(8)))  float    v8f;

#define NB 128   // batch
#define NC 64    // channels
#define NH 32
#define NW 32
#define NHP 34   // padded (halo) height
#define NWP 34   // padded (halo) width
#define NEDGE 6
#define KSTEPS 18                        // 9 taps * 64 ci / 32-K-per-wmma
#define FRAG_PER_EDGE (4*KSTEPS*32*16)   // halves: co-tiles * ksteps * lanes * 16
#define IMGP ((size_t)NB * NHP * NWP * NC)   // halves per padded image buffer

// ---------------------------------------------------------------------------
// Kernel 0: fold channel-scale into weights, convert to f16, and store in
// WMMA A-fragment order so the conv kernel's A load is one contiguous 32B/lane.
// Fragment index = ((((l*4 + cot)*18 + ks)*32 + lane)*16 + j)
// A fragment in this code's contiguous K order, same as B (16x32): lane<16 holds K = 0..15 (j=0..15);
// lane>=16 holds K = 16..31.  K maps to ci within a
// 32-wide ci block selected by (ks & 1); tap = ks >> 1.
// ---------------------------------------------------------------------------
__global__ void prep_weights(const float* __restrict__ W,
                             const float* __restrict__ a1,
                             const float* __restrict__ a2,
                             _Float16* __restrict__ wfrag) {
  int idx   = blockIdx.x * 256 + threadIdx.x;        // < 6*4*18*32*16 = 221184
  int j     = idx & 15;
  int lane  = (idx >> 4) & 31;
  int ks    = (idx >> 9) % KSTEPS;
  int cot   = (idx / (512 * KSTEPS)) & 3;
  int l     = idx / (512 * KSTEPS * 4);
  int tap   = ks >> 1;
  int khalf = lane >> 4;
  int m     = lane & 15;
  int chunk = j >> 3;
  int jj    = j & 7;
  int k32   = khalf * 16 + chunk * 8 + jj;           // K within this 32-step
  int ci    = (ks & 1) * 32 + k32;
  int co    = cot * 16 + m;
  // cumulative alpha sums: a_in[ci] = sum_{i>=ci/8} alphas1[i]
  float ain = 0.f, aout = 0.f;
  for (int i = ci >> 3; i < 8; ++i) ain  += a1[i];
  for (int i = co >> 3; i < 8; ++i) aout += a2[i];
  float w = W[(((l * NC + co) * NC + ci) * 9) + tap] * ain * aout;
  wfrag[idx] = (_Float16)w;
}

// ---------------------------------------------------------------------------
// Kernel 1: zero the 1-pixel halo border of all three padded NHWC f16 buffers
// (x0, n1, n2 contiguous with stride IMGP). 132 halo pixels per 34x34 image.
// Run every call so graph replay is deterministic after d_ws poisoning.
// ---------------------------------------------------------------------------
__global__ void zero_halo(_Float16* __restrict__ base) {
  int idx = blockIdx.x * 256 + threadIdx.x;          // < 3*128*132*64
  int ci  = idx & 63;
  int t   = (idx >> 6) % 132;
  int n   = ((idx >> 6) / 132) % NB;
  int buf = (idx >> 6) / (132 * NB);
  int h, w;
  if (t < 34)       { h = 0;          w = t;        }
  else if (t < 68)  { h = 33;         w = t - 34;   }
  else if (t < 100) { h = t - 68 + 1; w = 0;        }
  else              { h = t - 100 + 1; w = 33;      }
  base[(size_t)buf * IMGP + (((size_t)n * NHP + h) * NWP + w) * NC + ci] =
      (_Float16)0.f;
}

// ---------------------------------------------------------------------------
// Kernel 2: relu(x) and NCHW f32 -> padded NHWC f16 via LDS transpose.
// One block per (n,h) row: coalesced reads over w, coalesced writes over ci.
// ---------------------------------------------------------------------------
__global__ void prep_input(const float* __restrict__ x, _Float16* __restrict__ y) {
  __shared__ _Float16 tile[NW][NC + 4];
  int n = blockIdx.x >> 5;
  int h = blockIdx.x & 31;
  int t = threadIdx.x;
  int w0 = t & 31, cg = t >> 5;                      // cg in 0..7
  #pragma unroll
  for (int s = 0; s < 8; ++s) {
    int ci = s * 8 + cg;
    float v = x[((n * NC + ci) * NH + h) * NW + w0];
    tile[w0][ci] = (_Float16)fmaxf(v, 0.f);
  }
  __syncthreads();
  int ci2 = t & 63, wg = t >> 6;                     // wg in 0..3
  #pragma unroll
  for (int s = 0; s < 8; ++s) {
    int w = wg * 8 + s;
    y[(((size_t)n * NHP + (h + 1)) * NWP + (w + 1)) * NC + ci2] = tile[w][ci2];
  }
}

// ---------------------------------------------------------------------------
// Conv kernel: implicit-GEMM 3x3 SAME conv over 1..3 halo-padded sources.
// Each wave computes a 16co x 32pixel strip (one full image row) as TWO
// independent WMMA accumulator chains that share each A fragment:
// per k-step = 2 A-loads + 4 B-loads + 2 v_wmma (3 loads/WMMA, dual chains).
// Block = 8 waves covering 2 rows x 4 co-tiles; all B displacements are
// compile-time instruction offsets (halo padding, no predication).
// ---------------------------------------------------------------------------
template <int NSRC, bool RELU_F16_OUT>
__global__ void __launch_bounds__(256)
conv_cell(const _Float16* __restrict__ s0, const _Float16* __restrict__ s1,
          const _Float16* __restrict__ s2,
          const _Float16* __restrict__ wf0, const _Float16* __restrict__ wf1,
          const _Float16* __restrict__ wf2,
          _Float16* __restrict__ out16, float* __restrict__ out32) {
  int n    = blockIdx.x >> 4;          // image
  int hp   = blockIdx.x & 15;          // row pair
  int wave = threadIdx.x >> 5;
  int lane = threadIdx.x & 31;
  int cot  = wave & 3;                 // which 16-wide co tile
  int hsub = wave >> 2;                // which row of the pair
  int h    = hp * 2 + hsub;
  int col  = lane & 15;                // B column / D column
  int khalf = lane >> 4;               // which K half-block this lane carries
  int wpix = col;                      // pixel group 0: w = col, group 1: w = col+16

  const _Float16* srcs[3] = {s0, s1, s2};
  const _Float16* wfs[3]  = {wf0, wf1, wf2};

  // one base address per (wave,lane); all tap/ci displacements are immediates
  size_t pixbase = (((size_t)n * NHP + (h + 1)) * NWP + (wpix + 1)) * NC;

  v8f acc0 = {};
  v8f acc1 = {};
  #pragma unroll
  for (int s = 0; s < NSRC; ++s) {
    const _Float16* __restrict__ src   = srcs[s] + pixbase + khalf * 16;
    const _Float16* __restrict__ wbase =
        wfs[s] + (size_t)cot * (KSTEPS * 512) + lane * 16;
    #pragma unroll
    for (int ks = 0; ks < KSTEPS; ++ks) {
      int tap = ks >> 1;
      int dh  = tap / 3 - 1;
      int dw  = tap % 3 - 1;
      int off = (dh * NWP + dw) * NC + (ks & 1) * 32;  // compile-time constant
      v16h a  = *(const v16h*)(wbase + ks * 512);
      v16h b0 = *(const v16h*)(src + off);
      v16h b1 = *(const v16h*)(src + off + 16 * NC);   // pixels w+16
      acc0 = __builtin_amdgcn_wmma_f32_16x16x32_f16(
          false, a, false, b0, (short)0, acc0, false, false);
      acc1 = __builtin_amdgcn_wmma_f32_16x16x32_f16(
          false, a, false, b1, (short)0, acc1, false, false);
    }
  }

  // D layout: VGPR r, lane -> M = khalf*8 + r, N = col
  int co0 = cot * 16 + khalf * 8;
  if constexpr (RELU_F16_OUT) {
    v8h o0, o1;
    #pragma unroll
    for (int r = 0; r < 8; ++r) {
      o0[r] = (_Float16)fmaxf(acc0[r], 0.f);
      o1[r] = (_Float16)fmaxf(acc1[r], 0.f);
    }
    _Float16* dst = out16 + (((size_t)n * NHP + (h + 1)) * NWP + (wpix + 1)) * NC + co0;
    *(v8h*)(dst) = o0;
    *(v8h*)(dst + 16 * NC) = o1;
  } else {
    #pragma unroll
    for (int r = 0; r < 8; ++r) {
      float* dst = out32 + (((size_t)n * NC + (co0 + r)) * NH + h) * NW + wpix;
      dst[0]  = acc0[r];
      dst[16] = acc1[r];
    }
  }
}

// ---------------------------------------------------------------------------
extern "C" void kernel_launch(void* const* d_in, const int* in_sizes, int n_in,
                              void* d_out, int out_size, void* d_ws, size_t ws_size,
                              hipStream_t stream) {
  const float* x  = (const float*)d_in[0];   // [128,64,32,32]
  const float* a1 = (const float*)d_in[1];   // [8]
  const float* a2 = (const float*)d_in[2];   // [8]
  const float* W  = (const float*)d_in[3];   // [6,64,64,3,3]
  float* out = (float*)d_out;

  char* ws = (char*)d_ws;
  _Float16* wfrag = (_Float16*)ws;                   // 442,368 B
  _Float16* x0 = (_Float16*)(ws + (1 << 20));        // padded relu(input) NHWC f16
  _Float16* n1 = x0 + IMGP;                          // padded relu(node1)
  _Float16* n2 = n1 + IMGP;                          // padded relu(node2)

  prep_weights<<<(NEDGE * FRAG_PER_EDGE) / 256, 256, 0, stream>>>(W, a1, a2, wfrag);
  zero_halo<<<(3 * NB * 132 * 64) / 256, 256, 0, stream>>>(x0);
  prep_input<<<NB * NH, 256, 0, stream>>>(x, x0);

  const size_t E = FRAG_PER_EDGE;
  // node1 = conv(x0, W0)
  conv_cell<1, true><<<NB * NH / 2, 256, 0, stream>>>(
      x0, nullptr, nullptr, wfrag + 0 * E, nullptr, nullptr, n1, nullptr);
  // node2 = conv(x0, W1) + conv(n1, W2)
  conv_cell<2, true><<<NB * NH / 2, 256, 0, stream>>>(
      x0, n1, nullptr, wfrag + 1 * E, wfrag + 2 * E, nullptr, n2, nullptr);
  // node3 = conv(x0, W3) + conv(n1, W4) + conv(n2, W5)  -> f32 NCHW output
  conv_cell<3, false><<<NB * NH / 2, 256, 0, stream>>>(
      x0, n1, n2, wfrag + 3 * E, wfrag + 4 * E, wfrag + 5 * E, nullptr, out);
}
